// MambaKANPredictor_82523501625838
// MI455X (gfx1250) — matrix operand order repaired and checked by reading, not compiled, not run on hardware
//
#include <hip/hip_runtime.h>
#include <hip/hip_bf16.h>

typedef __attribute__((ext_vector_type(16))) __bf16 v16bf;
typedef __attribute__((ext_vector_type(8)))  float  v8f;
typedef __attribute__((ext_vector_type(4)))  int    v4i;

typedef __attribute__((address_space(1))) v4i gv4i;  // global int4
typedef __attribute__((address_space(3))) v4i lv4i;  // LDS int4

#if defined(__gfx1250__) && __has_builtin(__builtin_amdgcn_global_load_async_to_lds_b128)
#define USE_ASYNC_LDS 1
#else
#define USE_ASYNC_LDS 0
#endif

#if __has_builtin(__builtin_amdgcn_s_wait_asynccnt)
#define WAIT_ASYNC() __builtin_amdgcn_s_wait_asynccnt(0)
#else
#define WAIT_ASYNC() asm volatile("s_wait_asynccnt 0x0" ::: "memory")
#endif

__device__ __forceinline__ __bf16 f2bf(float f) { return (__bf16)f; }

// A-fragment, contiguous order: lane half lh holds K = lh*16+0..15 (elems 0..15).
__device__ __forceinline__ v16bf afrag_bf(const __bf16* p, int lh) {
  v16bf r;
#pragma unroll
  for (int i = 0; i < 16; ++i) r[i] = p[lh * 16 + i];
  return r;
}

#define TILE_M 128
#define TILE_N 128
#define TILE_K 32
#define LDS_STRIDE 40   // bf16 elems; 80B pitch -> 16B aligned, bank-spread

// One 32B tile-slice copy per thread (two b128 transfers), async when available.
__device__ __forceinline__ void stage_slice(const __bf16* g, __bf16* s) {
#if USE_ASYNC_LDS
  __builtin_amdgcn_global_load_async_to_lds_b128((gv4i*)g, (lv4i*)s, 0, 0);
  __builtin_amdgcn_global_load_async_to_lds_b128((gv4i*)(g + 8), (lv4i*)(s + 8), 0, 0);
#else
  *(v16bf*)s = *(const v16bf*)g;
#endif
}

// C[M,N] = A[M,K] @ W[N,K]^T (+bias) (+resid); A,W are bf16; C fp32;
// optional bf16 copy of C (first bfN columns) for chaining into the next GEMM.
__global__ __launch_bounds__(256) void gemm128_bf16(
    const __bf16* __restrict__ A, int lda,
    const __bf16* __restrict__ W, int ldw,
    const float* __restrict__ bias,
    const float* __restrict__ resid, int ldr,
    float* __restrict__ C, int ldc,
    __bf16* __restrict__ Cbf, int ldcb, int bfN,
    int K)
{
  __shared__ __bf16 As[2][TILE_M * LDS_STRIDE];
  __shared__ __bf16 Bs[2][TILE_N * LDS_STRIDE];

  const int m0 = blockIdx.y * TILE_M;
  const int n0 = blockIdx.x * TILE_N;
  const int tid = threadIdx.x;
  const int lane = tid & 31;
  const int wave = tid >> 5;
  const int wm = (wave >> 1) * 32;   // 4 waves along M
  const int wn = (wave & 1) * 64;    // 2 waves along N
  const int lr = lane & 15;
  const int lh = lane >> 4;

  const int ldRow = tid >> 1;        // 0..127
  const int ldSeg = (tid & 1) * 16;  // 0 or 16 (bf16 elems)

  const __bf16* gA = A + (size_t)(m0 + ldRow) * lda + ldSeg;
  const __bf16* gW = W + (size_t)(n0 + ldRow) * ldw + ldSeg;
  __bf16* sAbase = &As[0][ldRow * LDS_STRIDE + ldSeg];
  __bf16* sBbase = &Bs[0][ldRow * LDS_STRIDE + ldSeg];
  const int bufStride = TILE_M * LDS_STRIDE;

  v8f acc[2][4] = {};
  const int nk = K / TILE_K;

  // prologue: tile 0 -> buffer 0
  stage_slice(gA, sAbase);
  stage_slice(gW, sBbase);
#if USE_ASYNC_LDS
  WAIT_ASYNC();
#endif
  __syncthreads();

  for (int kt = 0; kt < nk; ++kt) {
    const int cur = kt & 1;
    if (kt + 1 < nk) {
      const int nxt = cur ^ 1;
      stage_slice(gA + (size_t)(kt + 1) * TILE_K, sAbase + nxt * bufStride);
      stage_slice(gW + (size_t)(kt + 1) * TILE_K, sBbase + nxt * bufStride);
    }

    v16bf af[2], bfr[4];
#pragma unroll
    for (int i = 0; i < 2; ++i)
      af[i] = afrag_bf(&As[cur][(wm + i * 16 + lr) * LDS_STRIDE], lh);
#pragma unroll
    for (int j = 0; j < 4; ++j)
      bfr[j] = *(const v16bf*)&Bs[cur][(wn + j * 16 + lr) * LDS_STRIDE + lh * 16];
#pragma unroll
    for (int i = 0; i < 2; ++i)
#pragma unroll
      for (int j = 0; j < 4; ++j)
        acc[i][j] = __builtin_amdgcn_wmma_f32_16x16x32_bf16(
            false, af[i], false, bfr[j], (short)0, acc[i][j], false, false);

#if USE_ASYNC_LDS
    WAIT_ASYNC();
#endif
    __syncthreads();
  }

#pragma unroll
  for (int i = 0; i < 2; ++i) {
#pragma unroll
    for (int j = 0; j < 4; ++j) {
      const int nn = n0 + wn + j * 16 + lr;
      const float bv = bias ? bias[nn] : 0.f;
#pragma unroll
      for (int v = 0; v < 8; ++v) {
        const int mm = m0 + wm + i * 16 + lh * 8 + v;
        float val = acc[i][j][v] + bv;
        if (resid) val += resid[(size_t)mm * ldr + nn];
        C[(size_t)mm * ldc + nn] = val;
        if (Cbf && nn < bfN) Cbf[(size_t)mm * ldcb + nn] = f2bf(val);
      }
    }
  }
}

// C[M,16] = A[M,K] @ W[16,K]^T  (Bx precompute), bf16 inputs.
__global__ __launch_bounds__(128) void gemm_n16_bf16(
    const __bf16* __restrict__ A, int lda,
    const __bf16* __restrict__ W,
    float* __restrict__ C, int K)
{
  const int lane = threadIdx.x & 31;
  const int wave = threadIdx.x >> 5;
  const int lr = lane & 15;
  const int lh = lane >> 4;
  const int m0 = blockIdx.x * 64 + wave * 16;
  v8f acc = {};
  for (int k0 = 0; k0 < K; k0 += 32) {
    v16bf af = afrag_bf(A + (size_t)(m0 + lr) * lda + k0, lh);
    v16bf bf_ = *(const v16bf*)(W + (size_t)lr * K + k0 + lh * 16);
    acc = __builtin_amdgcn_wmma_f32_16x16x32_bf16(
        false, af, false, bf_, (short)0, acc, false, false);
  }
#pragma unroll
  for (int v = 0; v < 8; ++v)
    C[(size_t)(m0 + lh * 8 + v) * 16 + lr] = acc[v];
}

// Sequential 16-state recurrence: h_t = tanh(h_{t-1}@A + Bx_t); store h_t (bf16).
__global__ __launch_bounds__(64) void ssm_scan(
    const float* __restrict__ A, const float* __restrict__ BX,
    __bf16* __restrict__ HSbf)
{
  const int lane = threadIdx.x & 31;
  const int wave = threadIdx.x >> 5;
  const int j = lane & 15;
  const int batch = blockIdx.x * 4 + wave * 2 + (lane >> 4);
  float a[16];
#pragma unroll
  for (int k = 0; k < 16; ++k) a[k] = A[k * 16 + j];
  const float* bx = BX + (size_t)batch * 512 * 16;
  __bf16* hs = HSbf + (size_t)batch * 512 * 16;
  float h = 0.f;
  for (int t = 0; t < 512; ++t) {
    float accv = bx[t * 16 + j];
#pragma unroll
    for (int k = 0; k < 16; ++k) accv += __shfl(h, k, 16) * a[k];
    h = tanhf(accv);
    hs[t * 16 + j] = f2bf(h);
  }
}

// y = (HS @ Cm^T + x_ssm*D) * sigmoid(gate); y written bf16 into XSbf.
// K=16, zero-padded to 32 in the WMMA fragments.
__global__ __launch_bounds__(256) void ssm_out_gemm(
    const __bf16* __restrict__ HSbf,
    const __bf16* __restrict__ Cmbf,
    const float* __restrict__ Dv,
    const float* __restrict__ XP,    // fp32 [M,1024]: x_ssm | gate
    __bf16* __restrict__ XSbf)       // [M,512] bf16, overwritten with y
{
  const int lane = threadIdx.x & 31;
  const int wave = threadIdx.x >> 5;
  const int lr = lane & 15;
  const int lh = lane >> 4;
  const int m0 = blockIdx.y * 128 + wave * 16;
  const int n0 = blockIdx.x * 64;

  const __bf16 zb = f2bf(0.f);
  v16bf af;
  {
    const __bf16* p = HSbf + (size_t)(m0 + lr) * 16;
#pragma unroll
    for (int i = 0; i < 16; ++i) af[i] = lh == 0 ? p[i] : zb;  // K=0..15 in lh=0, K=16..31 pad in lh=1
  }

  v8f acc[4] = {};
#pragma unroll
  for (int jt = 0; jt < 4; ++jt) {
    v16bf bf_;
    if (lh == 0) bf_ = *(const v16bf*)(Cmbf + (size_t)(n0 + jt * 16 + lr) * 16);
    else {
#pragma unroll
      for (int i = 0; i < 16; ++i) bf_[i] = zb;
    }
    acc[jt] = __builtin_amdgcn_wmma_f32_16x16x32_bf16(
        false, af, false, bf_, (short)0, acc[jt], false, false);
  }

#pragma unroll
  for (int jt = 0; jt < 4; ++jt) {
    const int nn = n0 + jt * 16 + lr;
    const float dv = Dv[nn];
#pragma unroll
    for (int v = 0; v < 8; ++v) {
      const int mm = m0 + lh * 8 + v;
      const size_t rowb = (size_t)mm * 1024;
      const float xs = XP[rowb + nn];
      const float gt = XP[rowb + 512 + nn];
      const float y = (acc[jt][v] + xs * dv) * (1.f / (1.f + __expf(-gt)));
      XSbf[(size_t)mm * 512 + nn] = f2bf(y);
    }
  }
}

__global__ __launch_bounds__(256) void layernorm_kernel(
    const float* __restrict__ X, const float* __restrict__ g,
    const float* __restrict__ b, float* __restrict__ O,
    __bf16* __restrict__ Obf)
{
  __shared__ float rs[8], rq[8];
  const int row = blockIdx.x;
  const float* x = X + (size_t)row * 512;
  const float v0 = x[threadIdx.x];
  const float v1 = x[threadIdx.x + 256];
  float s = v0 + v1;
  float q = v0 * v0 + v1 * v1;
  const int lane = threadIdx.x & 31, w = threadIdx.x >> 5;
#pragma unroll
  for (int off = 16; off > 0; off >>= 1) {
    s += __shfl_down(s, off, 32);
    q += __shfl_down(q, off, 32);
  }
  if (lane == 0) { rs[w] = s; rq[w] = q; }
  __syncthreads();
  float S = 0.f, Q = 0.f;
#pragma unroll
  for (int i = 0; i < 8; ++i) { S += rs[i]; Q += rq[i]; }
  const float mu = S * (1.f / 512.f);
  const float var = Q * (1.f / 512.f) - mu * mu;
  const float inv = rsqrtf(var + 1e-5f);
  const int c0 = threadIdx.x, c1 = threadIdx.x + 256;
  const float o0 = (v0 - mu) * inv * g[c0] + b[c0];
  const float o1 = (v1 - mu) * inv * g[c1] + b[c1];
  O[(size_t)row * 512 + c0] = o0;
  O[(size_t)row * 512 + c1] = o1;
  Obf[(size_t)row * 512 + c0] = f2bf(o0);
  Obf[(size_t)row * 512 + c1] = f2bf(o1);
}

__global__ __launch_bounds__(256) void convert_bf16(
    const float* __restrict__ in, __bf16* __restrict__ out, int n)
{
  const int i = (blockIdx.x * 256 + threadIdx.x) * 4;
  if (i < n) {
    const float4 v = *(const float4*)(in + i);
    out[i + 0] = f2bf(v.x);
    out[i + 1] = f2bf(v.y);
    out[i + 2] = f2bf(v.z);
    out[i + 3] = f2bf(v.w);
  }
}

// KAN heads + uncertainty on last timestep. One block per batch, 256 threads.
__global__ __launch_bounds__(256) void head_kernel(
    const float* __restrict__ Hf,
    const float* __restrict__ k1bw, const float* __restrict__ k1bb,
    const float* __restrict__ k1sw,
    const float* __restrict__ k2bw, const float* __restrict__ k2bb,
    const float* __restrict__ k2sw,
    const float* __restrict__ uw1, const float* __restrict__ ub1,
    const float* __restrict__ uw2, const float* __restrict__ ub2,
    float* __restrict__ out)
{
  __shared__ float last[512];
  __shared__ float basis[512 * 5];
  __shared__ float redp[8], redq[8];
  const int b = blockIdx.x, t = threadIdx.x;
  const float* src = Hf + ((size_t)b * 512 + 511) * 512;
  for (int i = t; i < 512; i += 256) {
    const float v = src[i];
    last[i] = v;
    const float xc = fminf(1.f, fmaxf(-1.f, v));
#pragma unroll
    for (int gi = 0; gi < 5; ++gi) {
      const float d = xc - (-1.f + 0.5f * gi);
      basis[i * 5 + gi] = __expf(-d * d);
    }
  }
  __syncthreads();

  float base = k1bb[t];
  float u = ub1[t];
  const float* wrow = k1bw + (size_t)t * 512;
  const float* urow = uw1 + (size_t)t * 512;
  const float* srow = k1sw + (size_t)t * 2560;
  for (int i = 0; i < 512; ++i) {
    const float li = last[i];
    base += li * wrow[i];
    u += li * urow[i];
  }
  float spline = 0.f;
  for (int i = 0; i < 2560; ++i) spline += basis[i] * srow[i];
  const float k1v = fmaxf(0.f, base + spline);
  u = fmaxf(0.f, u);

  const float xc = fminf(1.f, fmaxf(-1.f, k1v));
  float p = k1v * k2bw[t];
#pragma unroll
  for (int gi = 0; gi < 5; ++gi) {
    const float d = xc - (-1.f + 0.5f * gi);
    p += __expf(-d * d) * k2sw[t * 5 + gi];
  }
  float q = u * uw2[t];
  const int lane = t & 31, w = t >> 5;
#pragma unroll
  for (int off = 16; off > 0; off >>= 1) {
    p += __shfl_down(p, off, 32);
    q += __shfl_down(q, off, 32);
  }
  if (lane == 0) { redp[w] = p; redq[w] = q; }
  __syncthreads();
  if (t == 0) {
    float P = 0.f, Q = 0.f;
#pragma unroll
    for (int i = 0; i < 8; ++i) { P += redp[i]; Q += redq[i]; }
    out[b] = P + k2bb[0];
    const float z = Q + ub2[0];
    out[64 + b] = logf(1.f + __expf(z));
  }
}

extern "C" void kernel_launch(void* const* d_in, const int* in_sizes, int n_in,
                              void* d_out, int out_size, void* d_ws, size_t ws_size,
                              hipStream_t stream) {
  const float* x     = (const float*)d_in[0];
  const float* emb_w = (const float*)d_in[1];
  const float* emb_b = (const float*)d_in[2];
  const float* in_w  = (const float*)d_in[3];
  const float* in_b  = (const float*)d_in[4];
  const float* Amat  = (const float*)d_in[5];
  const float* Bm    = (const float*)d_in[6];
  const float* Cmt   = (const float*)d_in[7];
  const float* Dm    = (const float*)d_in[8];
  const float* out_w = (const float*)d_in[9];
  const float* out_b = (const float*)d_in[10];
  const float* ln_g  = (const float*)d_in[11];
  const float* ln_b  = (const float*)d_in[12];
  const float* k1bw  = (const float*)d_in[13];
  const float* k1bb  = (const float*)d_in[14];
  const float* k1sw  = (const float*)d_in[15];
  const float* k2bw  = (const float*)d_in[16];
  const float* k2bb  = (const float*)d_in[17];
  const float* k2sw  = (const float*)d_in[18];
  const float* uw1   = (const float*)d_in[19];
  const float* ub1   = (const float*)d_in[20];
  const float* uw2   = (const float*)d_in[21];
  const float* ub2   = (const float*)d_in[22];
  float* out = (float*)d_out;

  const int M = 64 * 512;  // 32768 rows
  char* p = (char*)d_ws;
  // fp32 buffers
  float* H0 = (float*)p; p += (size_t)M * 512 * 4;
  float* H1 = (float*)p; p += (size_t)M * 512 * 4;
  float* XP = (float*)p; p += (size_t)M * 1024 * 4;
  float* BX = (float*)p; p += (size_t)M * 16 * 4;
  float* R  = (float*)p; p += (size_t)M * 512 * 4;
  // bf16 buffers
  __bf16* Hbf  = (__bf16*)p; p += (size_t)M * 512 * 2;
  __bf16* XSbf = (__bf16*)p; p += (size_t)M * 512 * 2;
  __bf16* HSbf = (__bf16*)p; p += (size_t)M * 16 * 2;
  __bf16* xbf  = (__bf16*)p; p += (size_t)M * 32 * 2;
  __bf16* embw_bf = (__bf16*)p; p += (size_t)512 * 32 * 2;
  __bf16* inw_bf  = (__bf16*)p; p += (size_t)4 * 1024 * 512 * 2;
  __bf16* outw_bf = (__bf16*)p; p += (size_t)4 * 512 * 512 * 2;
  __bf16* bm_bf   = (__bf16*)p; p += (size_t)4 * 16 * 512 * 2;
  __bf16* cm_bf   = (__bf16*)p; p += (size_t)4 * 512 * 16 * 2;

  dim3 blk(256);
  // One-time (per call) weight / input conversions to bf16
  convert_bf16<<<dim3(M * 32 / 1024), blk, 0, stream>>>(x, xbf, M * 32);
  convert_bf16<<<dim3(512 * 32 / 1024), blk, 0, stream>>>(emb_w, embw_bf, 512 * 32);
  convert_bf16<<<dim3(4 * 1024 * 512 / 1024), blk, 0, stream>>>(in_w, inw_bf, 4 * 1024 * 512);
  convert_bf16<<<dim3(4 * 512 * 512 / 1024), blk, 0, stream>>>(out_w, outw_bf, 4 * 512 * 512);
  convert_bf16<<<dim3(4 * 16 * 512 / 1024), blk, 0, stream>>>(Bm, bm_bf, 4 * 16 * 512);
  convert_bf16<<<dim3(4 * 512 * 16 / 1024), blk, 0, stream>>>(Cmt, cm_bf, 4 * 512 * 16);

  // Embedding: H0 = x @ emb_w^T + emb_b (K=32); also bf16 copy Hbf
  gemm128_bf16<<<dim3(512 / 128, M / 128), blk, 0, stream>>>(
      xbf, 32, embw_bf, 32, emb_b, nullptr, 0, H0, 512, Hbf, 512, 512, 32);

  float* Hc = H0;
  float* Hn = H1;
  for (int l = 0; l < 4; ++l) {
    const __bf16* wl  = inw_bf + (size_t)l * 1024 * 512;
    const float*  bl  = in_b + (size_t)l * 1024;
    const float*  Al  = Amat + (size_t)l * 256;
    const __bf16* Bml = bm_bf + (size_t)l * 16 * 512;
    const __bf16* Cml = cm_bf + (size_t)l * 512 * 16;
    const float*  Dl  = Dm + (size_t)l * 512;
    const __bf16* owl = outw_bf + (size_t)l * 512 * 512;
    const float*  obl = out_b + (size_t)l * 512;

    // in_proj: XP = Hbf @ in_w^T + in_b; bf16 copy of x_ssm half into XSbf
    gemm128_bf16<<<dim3(1024 / 128, M / 128), blk, 0, stream>>>(
        Hbf, 512, wl, 512, bl, nullptr, 0, XP, 1024, XSbf, 512, 512, 512);
    // Bx = x_ssm @ Bm^T
    gemm_n16_bf16<<<dim3(M / 64), dim3(128), 0, stream>>>(XSbf, 512, Bml, BX, 512);
    // sequential recurrence over S (writes bf16 states)
    ssm_scan<<<dim3(16), dim3(64), 0, stream>>>(Al, BX, HSbf);
    // y = (HS @ Cm^T + x_ssm*D) * sigmoid(gate) -> XSbf (bf16)
    ssm_out_gemm<<<dim3(512 / 64, M / 128), blk, 0, stream>>>(HSbf, Cml, Dl, XP, XSbf);
    // out_proj + residual: R = y @ out_w^T + out_b + Hc
    gemm128_bf16<<<dim3(512 / 128, M / 128), blk, 0, stream>>>(
        XSbf, 512, owl, 512, obl, Hc, 512, R, 512, nullptr, 0, 0, 512);
    // layernorm -> Hn (fp32) + Hbf (bf16 for next layer's GEMM)
    layernorm_kernel<<<dim3(M), blk, 0, stream>>>(
        R, ln_g + l * 512, ln_b + l * 512, Hn, Hbf);
    float* tmp = Hc; Hc = Hn; Hn = tmp;
  }

  head_kernel<<<dim3(64), blk, 0, stream>>>(
      Hc, k1bw, k1bb, k1sw, k2bw, k2bb, k2sw, uw1, ub1, uw2, ub2, out);
}
